// MetapathAggregator_80307298501440
// MI455X (gfx1250) — matrix operand order repaired and checked by reading, not compiled, not run on hardware
//
#include <hip/hip_runtime.h>
#include <hip/hip_bf16.h>
#include <math.h>

typedef __attribute__((ext_vector_type(16))) __bf16 v16bf;
typedef __attribute__((ext_vector_type(8)))  __bf16 v8bf;   // 16 bytes = one b128 load
typedef __attribute__((ext_vector_type(8)))  float  v8f;

#define D_FEAT 128
#define NHEAD  8
#define CDIM   16
#define NEG_SLOPE 0.2f
#define LN_EPS 1e-5f

// ---------------------------------------------------------------------------
// Fill kernel (workspace init: -inf for segment-max bufs, 0 for accumulators)
// ---------------------------------------------------------------------------
__global__ void fill_f32(float* __restrict__ p, float v, size_t n) {
  size_t i = (size_t)blockIdx.x * blockDim.x + threadIdx.x;
  if (i < n) p[i] = v;
}

// ---------------------------------------------------------------------------
// Pre-conversion: X (f32, row-major) -> bf16 row-major
// ---------------------------------------------------------------------------
__global__ void conv_x_bf16(const float* __restrict__ X, __bf16* __restrict__ Xb,
                            size_t n) {
  size_t i = (size_t)blockIdx.x * blockDim.x + threadIdx.x;
  if (i < n) Xb[i] = (__bf16)X[i];
}

// Pre-conversion: W (f32, [K=128][N=128] row-major) -> bf16 transposed Wt[n][k]
__global__ void conv_w_bf16t(const float* __restrict__ Wm, __bf16* __restrict__ Wt) {
  int i = blockIdx.x * blockDim.x + threadIdx.x;   // 128*128 threads
  int k = i >> 7, n = i & 127;
  Wt[((size_t)n << 7) + k] = (__bf16)Wm[((size_t)k << 7) + n];
}

// ---------------------------------------------------------------------------
// H = X @ W via v_wmma_f32_16x16x32_bf16.  One wave -> one 16x16 output tile.
// Xb: [M,128] bf16 row-major; Wt: [128,128] bf16 (Wt[n][k] = W[k][n]).
// Inner loop: 4 x global_load_b128 + 1 WMMA, no conversions.
// Block = (32,4): 4 waves, each a different 16-col tile. grid=(ceil(M/16), 2).
// ---------------------------------------------------------------------------
__global__ __launch_bounds__(128) void gat_gemm_wmma(
    const __bf16* __restrict__ Xb, const __bf16* __restrict__ Wt,
    float* __restrict__ Hout, int M)
{
  const int lane    = threadIdx.x;   // 0..31
  const int halfsel = lane >> 4;     // 0 or 1 (half-wave)
  const int lm      = lane & 15;
  const int row0    = blockIdx.x * 16;
  const int col0    = (blockIdx.y * 4 + threadIdx.y) * 16;

  const int arow = min(row0 + lm, M - 1);          // clamp for ragged tail
  const __bf16* __restrict__ xr = Xb + ((size_t)arow << 7);
  const __bf16* __restrict__ wr = Wt + ((size_t)(col0 + lm) << 7); // col of W

  v8f acc = {};
#pragma unroll
  for (int k0 = 0; k0 < D_FEAT; k0 += 32) {
    // A fragment (16x32 bf16, ISA layout):
    //  lanes 0-15 : elems 0..7 = A[m][k0+0..7],  8..15 = A[m][k0+16..23]
    //  lanes 16-31: elems 0..7 = A[m][k0+8..15], 8..15 = A[m][k0+24..31]
    v8bf a0 = *(const v8bf*)(xr + k0 + halfsel * 8);
    v8bf a1 = *(const v8bf*)(xr + k0 + 16 + halfsel * 8);
    // B fragment (32x16 bf16), same k order as A: Wt[col0+lm][k]
    //  lanes 0-15 : elems 0..7 = k0+0..7,  8..15 = k0+16..23
    //  lanes 16-31: elems 0..7 = k0+8..15, 8..15 = k0+24..31
    v8bf b0 = *(const v8bf*)(wr + k0 + halfsel * 8);
    v8bf b1 = *(const v8bf*)(wr + k0 + 16 + halfsel * 8);

    v16bf a = __builtin_shufflevector(a0, a1, 0,1,2,3,4,5,6,7,8,9,10,11,12,13,14,15);
    v16bf b = __builtin_shufflevector(b0, b1, 0,1,2,3,4,5,6,7,8,9,10,11,12,13,14,15);

    acc = __builtin_amdgcn_wmma_f32_16x16x32_bf16(
        /*neg_a=*/false, a, /*neg_b=*/false, b,
        /*c_mod=*/(short)0, acc, /*reuse_a=*/false, /*reuse_b=*/false);
  }
  // D layout: VGPR r -> row (r + 8*halfsel), col lm.
  // Fast path (uniform): full 16-row tile -> unguarded store clause.
  float* __restrict__ hb = Hout + ((size_t)(row0 + 8 * halfsel) << 7) + col0 + lm;
  if (row0 + 16 <= M) {
#pragma unroll
    for (int r = 0; r < 8; ++r) hb[(size_t)r << 7] = acc[r];
  } else {
#pragma unroll
    for (int r = 0; r < 8; ++r) {
      if (row0 + r + 8 * halfsel < M) hb[(size_t)r << 7] = acc[r];
    }
  }
}

// ---------------------------------------------------------------------------
// al[n,h] = sum_c H[n, h*16+c] * a[h,c]     (one thread per (node,head))
// ---------------------------------------------------------------------------
__global__ void attn_logits(const float* __restrict__ Hm,
                            const float* __restrict__ av,
                            float* __restrict__ al, int N)
{
  int i = blockIdx.x * blockDim.x + threadIdx.x;
  if (i >= N * NHEAD) return;
  int node = i >> 3, h = i & 7;
  const float* hr = Hm + ((size_t)node << 7) + h * CDIM;
  const float* ar = av + h * CDIM;
  float s = 0.f;
#pragma unroll
  for (int c = 0; c < CDIM; ++c) s += hr[c] * ar[c];
  al[i] = s;
}

// ---------------------------------------------------------------------------
// Signed float atomic max via integer atomics (mbuf initialized to -inf).
// ---------------------------------------------------------------------------
__device__ __forceinline__ void atomicMaxF32(float* addr, float val) {
  if (val >= 0.f) atomicMax((int*)addr, __float_as_int(val));
  else            atomicMin((unsigned int*)addr, __float_as_uint(val));
}

__device__ __forceinline__ float leaky(float v) {
  return v > 0.f ? v : NEG_SLOPE * v;
}

// Pass A: segment max of leaky_relu(al_s[src]+al_d[dst]) per (dst, head)
__global__ void edge_seg_max(const int* __restrict__ src, const int* __restrict__ dst,
                             const float* __restrict__ als, const float* __restrict__ ald,
                             float* __restrict__ mbuf, int E)
{
  int i = blockIdx.x * blockDim.x + threadIdx.x;
  if (i >= E * NHEAD) return;
  int e = i >> 3, h = i & 7;
  int s = src[e], d = dst[e];
  float v = leaky(als[s * NHEAD + h] + ald[d * NHEAD + h]);
  atomicMaxF32(&mbuf[d * NHEAD + h], v);
}

// Pass B: denom[dst,h] += exp(e - m[dst,h])
__global__ void edge_seg_sum(const int* __restrict__ src, const int* __restrict__ dst,
                             const float* __restrict__ als, const float* __restrict__ ald,
                             const float* __restrict__ mbuf, float* __restrict__ denom, int E)
{
  int i = blockIdx.x * blockDim.x + threadIdx.x;
  if (i >= E * NHEAD) return;
  int e = i >> 3, h = i & 7;
  int s = src[e], d = dst[e];
  float v = leaky(als[s * NHEAD + h] + ald[d * NHEAD + h]);
  float p = expf(v - mbuf[d * NHEAD + h]);
  atomicAdd(&denom[d * NHEAD + h], p);
}

// Pass C: acc[dst, h*16+c] += alpha * Hsrc[src, h*16+c]
__global__ void edge_scatter(const int* __restrict__ src, const int* __restrict__ dst,
                             const float* __restrict__ als, const float* __restrict__ ald,
                             const float* __restrict__ mbuf, const float* __restrict__ denom,
                             const float* __restrict__ Hsrc, float* __restrict__ acc, int E)
{
  int i = blockIdx.x * blockDim.x + threadIdx.x;
  if (i >= E * NHEAD) return;
  int e = i >> 3, h = i & 7;
  int s = src[e], d = dst[e];
  float v = leaky(als[s * NHEAD + h] + ald[d * NHEAD + h]);
  float p = expf(v - mbuf[d * NHEAD + h]);
  float alpha = p / (denom[d * NHEAD + h] + 1e-16f);
  const float* hs = Hsrc + ((size_t)s << 7) + h * CDIM;
  float*       ap = acc  + ((size_t)d << 7) + h * CDIM;
#pragma unroll
  for (int c = 0; c < CDIM; ++c) atomicAdd(&ap[c], alpha * hs[c]);
}

// ---------------------------------------------------------------------------
// out = LayerNorm(x + 0.5*(acc + b1 + b2)) ; one wave per row, lane -> 4 cols
// ---------------------------------------------------------------------------
__global__ __launch_bounds__(128) void finalize_ln(
    const float* __restrict__ x, const float* __restrict__ acc,
    const float* __restrict__ b1, const float* __restrict__ b2,
    const float* __restrict__ lns, const float* __restrict__ lnb,
    float* __restrict__ out, int N)
{
  int row  = blockIdx.x * 4 + (threadIdx.x >> 5);
  int lane = threadIdx.x & 31;
  if (row >= N) return;
  const float* xr = x   + ((size_t)row << 7);
  const float* ar = acc + ((size_t)row << 7);

  float vals[4];
  float sum = 0.f;
#pragma unroll
  for (int j = 0; j < 4; ++j) {
    int c = lane * 4 + j;
    float t = xr[c] + 0.5f * (ar[c] + b1[c] + b2[c]);
    vals[j] = t; sum += t;
  }
#pragma unroll
  for (int off = 16; off >= 1; off >>= 1) sum += __shfl_xor(sum, off, 32);
  float mu = sum * (1.f / D_FEAT);

  float var = 0.f;
#pragma unroll
  for (int j = 0; j < 4; ++j) { float d = vals[j] - mu; var += d * d; }
#pragma unroll
  for (int off = 16; off >= 1; off >>= 1) var += __shfl_xor(var, off, 32);
  var *= (1.f / D_FEAT);
  float rs = rsqrtf(var + LN_EPS);

  float* orow = out + ((size_t)row << 7);
#pragma unroll
  for (int j = 0; j < 4; ++j) {
    int c = lane * 4 + j;
    orow[c] = (vals[j] - mu) * rs * lns[c] + lnb[c];
  }
}

// ---------------------------------------------------------------------------
extern "C" void kernel_launch(void* const* d_in, const int* in_sizes, int n_in,
                              void* d_out, int out_size, void* d_ws, size_t ws_size,
                              hipStream_t stream) {
  const float* x_c   = (const float*)d_in[0];
  const float* x_i   = (const float*)d_in[1];
  const int*   ei_sp = (const int*)d_in[2];
  const int*   ei_bt = (const int*)d_in[3];
  const int*   ei_ct = (const int*)d_in[4];
  const int*   ei_sc = (const int*)d_in[5];
  const float* W_sp = (const float*)d_in[6],  *as_sp = (const float*)d_in[7],
             * ad_sp = (const float*)d_in[8], *b_sp  = (const float*)d_in[9];
  const float* W_bt = (const float*)d_in[10], *as_bt = (const float*)d_in[11],
             * ad_bt = (const float*)d_in[12],*b_bt  = (const float*)d_in[13];
  const float* W_ct = (const float*)d_in[14], *as_ct = (const float*)d_in[15],
             * ad_ct = (const float*)d_in[16],*b_ct  = (const float*)d_in[17];
  const float* W_sc = (const float*)d_in[18], *as_sc = (const float*)d_in[19],
             * ad_sc = (const float*)d_in[20],*b_sc  = (const float*)d_in[21];
  const float* ln_c_s = (const float*)d_in[22], *ln_c_b = (const float*)d_in[23];
  const float* ln_i_s = (const float*)d_in[24], *ln_i_b = (const float*)d_in[25];

  const int NC  = in_sizes[0] / D_FEAT;
  const int NI  = in_sizes[1] / D_FEAT;
  const int Esp = in_sizes[2] / 2, Ebt = in_sizes[3] / 2;
  const int Ect = in_sizes[4] / 2, Esc = in_sizes[5] / 2;
  const int* src_sp = ei_sp;            const int* dst_sp = ei_sp + Esp;
  const int* src_bt = ei_bt;            const int* dst_bt = ei_bt + Ebt;
  const int* src_ct = ei_ct;            const int* dst_ct = ei_ct + Ect;
  const int* src_sc = ei_sc;            const int* dst_sc = ei_sc + Esc;

  // ---- workspace carve-out ----
  float* w = (float*)d_ws;
  auto take = [&](size_t n) { float* p = w; w += n; return p; };
  float* h_c_sp = take((size_t)NC * D_FEAT);
  float* h_c_bt = take((size_t)NC * D_FEAT);
  float* h_c_ct = take((size_t)NC * D_FEAT);
  float* h_i_bt = take((size_t)NI * D_FEAT);
  float* h_i_ct = take((size_t)NI * D_FEAT);
  float* h_i_sc = take((size_t)NI * D_FEAT);
  float* al_s_sp = take((size_t)NC * NHEAD);
  float* al_d_sp = take((size_t)NC * NHEAD);
  float* al_s_bt = take((size_t)NC * NHEAD);
  float* al_d_ct = take((size_t)NC * NHEAD);
  float* al_d_bt = take((size_t)NI * NHEAD);
  float* al_s_ct = take((size_t)NI * NHEAD);
  float* al_s_sc = take((size_t)NI * NHEAD);
  float* al_d_sc = take((size_t)NI * NHEAD);
  // -inf region (contiguous)
  float* m_sp = take((size_t)NC * NHEAD);
  float* m_bt = take((size_t)NI * NHEAD);
  float* m_ct = take((size_t)NC * NHEAD);
  float* m_sc = take((size_t)NI * NHEAD);
  size_t n_minf = (size_t)(2 * NC + 2 * NI) * NHEAD;
  // zero region (contiguous)
  float* den_sp = take((size_t)NC * NHEAD);
  float* den_bt = take((size_t)NI * NHEAD);
  float* den_ct = take((size_t)NC * NHEAD);
  float* den_sc = take((size_t)NI * NHEAD);
  float* acc_c  = take((size_t)NC * D_FEAT);
  float* acc_i  = take((size_t)NI * D_FEAT);
  size_t n_zero = (size_t)(2 * NC + 2 * NI) * NHEAD
                + (size_t)NC * D_FEAT + (size_t)NI * D_FEAT;
  // bf16 region (all preceding counts are multiples of 8 -> 16B aligned)
  __bf16* bw = (__bf16*)w;
  auto takeb = [&](size_t n) { __bf16* p = bw; bw += n; return p; };
  __bf16* xbf_c = takeb((size_t)NC * D_FEAT);
  __bf16* xbf_i = takeb((size_t)NI * D_FEAT);
  __bf16* wt_sp = takeb((size_t)D_FEAT * D_FEAT);
  __bf16* wt_bt = takeb((size_t)D_FEAT * D_FEAT);
  __bf16* wt_ct = takeb((size_t)D_FEAT * D_FEAT);
  __bf16* wt_sc = takeb((size_t)D_FEAT * D_FEAT);

  float* out_c = (float*)d_out;
  float* out_i = out_c + (size_t)NC * D_FEAT;

  // ---- init + precision conversion ----
  fill_f32<<<(unsigned)((n_minf + 255) / 256), 256, 0, stream>>>(m_sp, -INFINITY, n_minf);
  fill_f32<<<(unsigned)((n_zero + 255) / 256), 256, 0, stream>>>(den_sp, 0.f, n_zero);

  size_t nxc = (size_t)NC * D_FEAT, nxi = (size_t)NI * D_FEAT;
  conv_x_bf16<<<(unsigned)((nxc + 255) / 256), 256, 0, stream>>>(x_c, xbf_c, nxc);
  conv_x_bf16<<<(unsigned)((nxi + 255) / 256), 256, 0, stream>>>(x_i, xbf_i, nxi);
  conv_w_bf16t<<<64, 256, 0, stream>>>(W_sp, wt_sp);
  conv_w_bf16t<<<64, 256, 0, stream>>>(W_bt, wt_bt);
  conv_w_bf16t<<<64, 256, 0, stream>>>(W_ct, wt_ct);
  conv_w_bf16t<<<64, 256, 0, stream>>>(W_sc, wt_sc);

  // ---- GEMMs (WMMA) ----
  dim3 gb(32, 4);
  dim3 gc((NC + 15) / 16, 2), gi((NI + 15) / 16, 2);
  gat_gemm_wmma<<<gc, gb, 0, stream>>>(xbf_c, wt_sp, h_c_sp, NC);
  gat_gemm_wmma<<<gc, gb, 0, stream>>>(xbf_c, wt_bt, h_c_bt, NC);
  gat_gemm_wmma<<<gc, gb, 0, stream>>>(xbf_c, wt_ct, h_c_ct, NC);
  gat_gemm_wmma<<<gi, gb, 0, stream>>>(xbf_i, wt_bt, h_i_bt, NI);
  gat_gemm_wmma<<<gi, gb, 0, stream>>>(xbf_i, wt_ct, h_i_ct, NI);
  gat_gemm_wmma<<<gi, gb, 0, stream>>>(xbf_i, wt_sc, h_i_sc, NI);

  // ---- attention logits ----
  int bc = (NC * NHEAD + 255) / 256, bi = (NI * NHEAD + 255) / 256;
  attn_logits<<<bc, 256, 0, stream>>>(h_c_sp, as_sp, al_s_sp, NC);
  attn_logits<<<bc, 256, 0, stream>>>(h_c_sp, ad_sp, al_d_sp, NC);
  attn_logits<<<bc, 256, 0, stream>>>(h_c_bt, as_bt, al_s_bt, NC);
  attn_logits<<<bi, 256, 0, stream>>>(h_i_bt, ad_bt, al_d_bt, NI);
  attn_logits<<<bi, 256, 0, stream>>>(h_i_ct, as_ct, al_s_ct, NI);
  attn_logits<<<bc, 256, 0, stream>>>(h_c_ct, ad_ct, al_d_ct, NC);
  attn_logits<<<bi, 256, 0, stream>>>(h_i_sc, as_sc, al_s_sc, NI);
  attn_logits<<<bi, 256, 0, stream>>>(h_i_sc, ad_sc, al_d_sc, NI);

  // ---- segment softmax: max, denom, scatter ----
  int esp = (Esp * NHEAD + 255) / 256, ebt = (Ebt * NHEAD + 255) / 256;
  int ect = (Ect * NHEAD + 255) / 256, esc = (Esc * NHEAD + 255) / 256;

  edge_seg_max<<<esp, 256, 0, stream>>>(src_sp, dst_sp, al_s_sp, al_d_sp, m_sp, Esp);
  edge_seg_max<<<ebt, 256, 0, stream>>>(src_bt, dst_bt, al_s_bt, al_d_bt, m_bt, Ebt);
  edge_seg_max<<<ect, 256, 0, stream>>>(src_ct, dst_ct, al_s_ct, al_d_ct, m_ct, Ect);
  edge_seg_max<<<esc, 256, 0, stream>>>(src_sc, dst_sc, al_s_sc, al_d_sc, m_sc, Esc);

  edge_seg_sum<<<esp, 256, 0, stream>>>(src_sp, dst_sp, al_s_sp, al_d_sp, m_sp, den_sp, Esp);
  edge_seg_sum<<<ebt, 256, 0, stream>>>(src_bt, dst_bt, al_s_bt, al_d_bt, m_bt, den_bt, Ebt);
  edge_seg_sum<<<ect, 256, 0, stream>>>(src_ct, dst_ct, al_s_ct, al_d_ct, m_ct, den_ct, Ect);
  edge_seg_sum<<<esc, 256, 0, stream>>>(src_sc, dst_sc, al_s_sc, al_d_sc, m_sc, den_sc, Esc);

  edge_scatter<<<esp, 256, 0, stream>>>(src_sp, dst_sp, al_s_sp, al_d_sp, m_sp, den_sp, h_c_sp, acc_c, Esp);
  edge_scatter<<<ebt, 256, 0, stream>>>(src_bt, dst_bt, al_s_bt, al_d_bt, m_bt, den_bt, h_c_bt, acc_i, Ebt);
  edge_scatter<<<ect, 256, 0, stream>>>(src_ct, dst_ct, al_s_ct, al_d_ct, m_ct, den_ct, h_i_ct, acc_c, Ect);
  edge_scatter<<<esc, 256, 0, stream>>>(src_sc, dst_sc, al_s_sc, al_d_sc, m_sc, den_sc, h_i_sc, acc_i, Esc);

  // ---- residual + bias mean + LayerNorm ----
  finalize_ln<<<(NC + 3) / 4, 128, 0, stream>>>(x_c, acc_c, b_sp, b_ct, ln_c_s, ln_c_b, out_c, NC);
  finalize_ln<<<(NI + 3) / 4, 128, 0, stream>>>(x_i, acc_i, b_bt, b_sc, ln_i_s, ln_i_b, out_i, NI);
}
